// GATLayer_13984413516295
// MI455X (gfx1250) — matrix operand order repaired and checked by reading, not compiled, not run on hardware
//
#include <hip/hip_runtime.h>
#include <hip/hip_bf16.h>

typedef __attribute__((ext_vector_type(16))) __bf16        v16bf;
typedef __attribute__((ext_vector_type(8)))  float         v8f;
typedef __attribute__((ext_vector_type(4)))  int           i32x4;
typedef __attribute__((ext_vector_type(4)))  float         f32x4;
typedef __attribute__((ext_vector_type(4)))  unsigned int  u32x4;

#define GAT_N     8192
#define GAT_FIN   128
#define GAT_FOUT  64
#define ALPHA     0.2f
#define LOG2E     1.44269504088896340736f
#define NCHUNK    4                 // j-dimension split for occupancy
#define CHUNK_J   (GAT_N / NCHUNK)  // 2048
#define NTILES    (GAT_N / 16)      // 512 row tiles
#define NWAVES    (NTILES * NCHUNK) // 2048 waves

__device__ __forceinline__ unsigned short f32_bf16(float f) {
  unsigned int u = __float_as_uint(f);
  u += 0x7fffu + ((u >> 16) & 1u);          // round-to-nearest-even
  return (unsigned short)(u >> 16);
}

__device__ __forceinline__ float fast_exp2(float x) {
#if __has_builtin(__builtin_amdgcn_exp2f)
  return __builtin_amdgcn_exp2f(x);         // v_exp_f32
#else
  return exp2f(x);
#endif
}

// ---------------------------------------------------------------------------
// Kernel 1: h = x @ W (one row per 64-thread block); emit bf16 transposed h
// (WMMA B-fragment friendly) and log2e-scaled logit vectors s1, s2.
// ---------------------------------------------------------------------------
__global__ __launch_bounds__(64) void k_hW(const float* __restrict__ x,
                                           const float* __restrict__ W,
                                           const float* __restrict__ a,
                                           unsigned short* __restrict__ hT,
                                           float* __restrict__ s1,
                                           float* __restrict__ s2) {
  const int i = blockIdx.x;
  const int f = threadIdx.x;
  const float* xr = x + (size_t)i * GAT_FIN;
  float acc = 0.f;
#pragma unroll 8
  for (int k = 0; k < GAT_FIN; ++k)
    acc = fmaf(xr[k], W[k * GAT_FOUT + f], acc);

  hT[(size_t)f * GAT_N + i] = f32_bf16(acc);

  __shared__ float r1[64];
  __shared__ float r2[64];
  r1[f] = acc * a[f];
  r2[f] = acc * a[GAT_FOUT + f];
  __syncthreads();
  for (int s = 32; s > 0; s >>= 1) {
    if (f < s) { r1[f] += r1[f + s]; r2[f] += r2[f + s]; }
    __syncthreads();
  }
  if (f == 0) { s1[i] = r1[0] * LOG2E; s2[i] = r2[0] * LOG2E; }  // log2 domain
}

// ---------------------------------------------------------------------------
// Kernel 2: s2max = max_j s2[j]  (fixed softmax shift; leakyrelu is monotone,
// log2e scaling is positive so max/leakyrelu commute with it)
// ---------------------------------------------------------------------------
__global__ __launch_bounds__(256) void k_s2max(const float* __restrict__ s2,
                                               float* __restrict__ s2max) {
  __shared__ float red[256];
  const int t = threadIdx.x;
  float m = -3.402823466e38f;
  for (int j = t; j < GAT_N; j += 256) m = fmaxf(m, s2[j]);
  red[t] = m;
  __syncthreads();
  for (int s = 128; s > 0; s >>= 1) {
    if (t < s) red[t] = fmaxf(red[t], red[t + s]);
    __syncthreads();
  }
  if (t == 0) *s2max = red[0];
}

// ---------------------------------------------------------------------------
// Kernel 3: fused mask -> exp2(shifted) -> attn-partial @ h via bf16 WMMA.
// One wave per (16-row tile, j-chunk). Copy-free modulo-4 software pipeline:
// depth-2 prefetch on the NT adj stream, depth-1 on L2-resident s2/hT tiles.
// ---------------------------------------------------------------------------
struct AdjTile { i32x4 a[4]; };
struct SBTile  { f32x4 s[4]; u32x4 b[8]; };

__device__ __forceinline__ AdjTile loadAdj(const int* __restrict__ adjRow,
                                           int j0, int kA) {
  AdjTile t;
  t.a[0] = __builtin_nontemporal_load((const i32x4*)(adjRow + j0 + kA));
  t.a[1] = __builtin_nontemporal_load((const i32x4*)(adjRow + j0 + kA + 4));
  t.a[2] = __builtin_nontemporal_load((const i32x4*)(adjRow + j0 + 16 + kA));
  t.a[3] = __builtin_nontemporal_load((const i32x4*)(adjRow + j0 + 16 + kA + 4));
  return t;
}

__device__ __forceinline__ SBTile loadSB(const float* __restrict__ s2,
                                         const unsigned short* const* hB,
                                         int j0, int kA) {
  SBTile t;
  t.s[0] = *(const f32x4*)(s2 + j0 + kA);
  t.s[1] = *(const f32x4*)(s2 + j0 + kA + 4);
  t.s[2] = *(const f32x4*)(s2 + j0 + 16 + kA);
  t.s[3] = *(const f32x4*)(s2 + j0 + 16 + kA + 4);
#pragma unroll
  for (int q = 0; q < 4; ++q) {
    t.b[2 * q]     = *(const u32x4*)(hB[q] + j0);
    t.b[2 * q + 1] = *(const u32x4*)(hB[q] + j0 + 16);
  }
  return t;
}

__global__ __launch_bounds__(256) void k_gat(const int* __restrict__ adj,
                                             const unsigned short* __restrict__ hT,
                                             const float* __restrict__ s1,
                                             const float* __restrict__ s2,
                                             const float* __restrict__ s2maxp,
                                             float* __restrict__ pacc,
                                             float* __restrict__ plsum) {
  const int lane   = threadIdx.x & 31;
  const int waveId = blockIdx.x * 8 + (threadIdx.x >> 5);
  const int chunk  = waveId >> 9;           // 0..3 (blocks never straddle)
  const int tile   = waveId & (NTILES - 1); // 0..511
  const int i0     = tile * 16;
  const int jBeg   = chunk * CHUNK_J;

  const int m  = lane & 15;   // A row id / B-C-D column id
  const int hi = lane >> 4;   // half-wave select
  const int kA = hi * 8;      // A-fragment K base (ISA 16-bit A layout)
  const int kB = hi * 8;      // B-fragment K base (ISA 16-bit B layout)
  const int row = i0 + m;

  const float s2max = *s2maxp;
  const float s1i   = s1[row];                    // already * log2e
  const float emax  = s1i + s2max;
  const float mi    = fmaxf(emax, ALPHA * emax);  // monotone upper bound shift

  const int* adjRow = adj + (size_t)row * GAT_N;
  const unsigned short* hB[4];
#pragma unroll
  for (int t = 0; t < 4; ++t)
    hB[t] = hT + (size_t)(16 * t + m) * GAT_N + kB;

  v8f acc[4];
#pragma unroll
  for (int t = 0; t < 4; ++t)
#pragma unroll
    for (int r = 0; r < 8; ++r) acc[t][r] = 0.f;
  float lp[4] = {0.f, 0.f, 0.f, 0.f};             // lsum partials (short chains)

  const int nIt = CHUNK_J / 32;                   // 64 iterations (16 groups)
  auto jAt = [&](int it) { return jBeg + ((it < nIt) ? it : 0) * 32; };

  auto computeTile = [&](const AdjTile& A, const SBTile& S) {
    union { v16bf v; unsigned short u[16]; } af;
#pragma unroll
    for (int e = 0; e < 16; ++e) {
      float ev = s1i + S.s[e >> 2][e & 3];
      ev = fmaxf(ev, ALPHA * ev);                 // leaky_relu (log2 domain)
      const float w =
          (A.a[e >> 2][e & 3] > 0) ? fast_exp2(ev - mi) : 0.f;  // arg <= 0
      lp[e >> 2] += w;
      af.u[e] = f32_bf16(w);
    }
#pragma unroll
    for (int t = 0; t < 4; ++t) {
      union { v16bf v; u32x4 q[2]; } bu;
      bu.q[0] = S.b[2 * t];
      bu.q[1] = S.b[2 * t + 1];
      acc[t] = __builtin_amdgcn_wmma_f32_16x16x32_bf16(
          false, af.v, false, bu.v, (short)0, acc[t], false, false);
    }
  };

  // Copy-free modulo pipeline: A buffers rotate with period 4 (depth-2
  // prefetch), S buffers with period 2 (depth-1). nIt % 4 == 0.
  AdjTile A0 = loadAdj(adjRow, jAt(0), kA);
  AdjTile A1 = loadAdj(adjRow, jAt(1), kA);
  AdjTile A2, A3;
  SBTile  S0 = loadSB(s2, hB, jAt(0), kA);
  SBTile  S1;

  for (int it = 0; it < nIt; it += 4) {
    A2 = loadAdj(adjRow, jAt(it + 2), kA);
    S1 = loadSB(s2, hB, jAt(it + 1), kA);
    computeTile(A0, S0);

    A3 = loadAdj(adjRow, jAt(it + 3), kA);
    S0 = loadSB(s2, hB, jAt(it + 2), kA);
    computeTile(A1, S1);

    A0 = loadAdj(adjRow, jAt(it + 4), kA);
    S1 = loadSB(s2, hB, jAt(it + 3), kA);
    computeTile(A2, S0);

    A1 = loadAdj(adjRow, jAt(it + 5), kA);
    S0 = loadSB(s2, hB, jAt(it + 4), kA);
    computeTile(A3, S1);
  }

  // lanes L and L^16 jointly cover all 32 K per tile for row (L&15)
  float lsum = (lp[0] + lp[1]) + (lp[2] + lp[3]);
  lsum += __shfl_xor(lsum, 16, 32);
  if (lane < 16) plsum[chunk * GAT_N + i0 + lane] = lsum;

#pragma unroll
  for (int r = 0; r < 8; ++r) {
    const int orow = r + hi * 8;                  // C/D row for this lane/VGPR
#pragma unroll
    for (int t = 0; t < 4; ++t)
      pacc[((size_t)chunk * GAT_N + i0 + orow) * GAT_FOUT + 16 * t + m] =
          acc[t][r];
  }
}

// ---------------------------------------------------------------------------
// Kernel 4: combine j-chunks, normalize (fast rcp), elu.
// ---------------------------------------------------------------------------
__global__ __launch_bounds__(256) void k_combine(const float* __restrict__ pacc,
                                                 const float* __restrict__ plsum,
                                                 float* __restrict__ out) {
  const int idx = blockIdx.x * 256 + threadIdx.x;   // < 8192*64
  const int row = idx >> 6;
  float s = 0.f, l = 0.f;
#pragma unroll
  for (int c = 0; c < NCHUNK; ++c) {
    s += pacc[(size_t)c * (GAT_N * GAT_FOUT) + idx];
    l += plsum[c * GAT_N + row];
  }
  const float hp = s * __builtin_amdgcn_rcpf(l);
  out[idx] = (hp > 0.f) ? hp : (fast_exp2(hp * LOG2E) - 1.f);   // elu
}

// ---------------------------------------------------------------------------
extern "C" void kernel_launch(void* const* d_in, const int* in_sizes, int n_in,
                              void* d_out, int out_size, void* d_ws, size_t ws_size,
                              hipStream_t stream) {
  const float* x   = (const float*)d_in[0];   // 8192 x 128
  const int*   adj = (const int*)d_in[1];     // 8192 x 8192
  const float* W   = (const float*)d_in[2];   // 128 x 64
  const float* a   = (const float*)d_in[3];   // 128 x 1
  float* out = (float*)d_out;                 // 8192 x 64

  char* ws = (char*)d_ws;
  const size_t off_hT    = 0;                              // 1 MB
  const size_t off_s1    = off_hT + (size_t)GAT_FOUT * GAT_N * 2;
  const size_t off_s2    = off_s1 + GAT_N * 4;
  const size_t off_s2m   = off_s2 + GAT_N * 4;
  const size_t off_plsum = off_s2m + 256;                  // 4*8192*4 = 128 KB
  const size_t off_pacc  = off_plsum + (size_t)NCHUNK * GAT_N * 4;  // 8 MB

  unsigned short* hT  = (unsigned short*)(ws + off_hT);
  float* s1    = (float*)(ws + off_s1);
  float* s2    = (float*)(ws + off_s2);
  float* s2m   = (float*)(ws + off_s2m);
  float* plsum = (float*)(ws + off_plsum);
  float* pacc  = (float*)(ws + off_pacc);

  k_hW     <<<dim3(GAT_N),          dim3(64),  0, stream>>>(x, W, a, hT, s1, s2);
  k_s2max  <<<dim3(1),              dim3(256), 0, stream>>>(s2, s2m);
  k_gat    <<<dim3(NWAVES / 8),     dim3(256), 0, stream>>>(adj, hT, s1, s2, s2m,
                                                            pacc, plsum);
  k_combine<<<dim3(GAT_N * GAT_FOUT / 256), dim3(256), 0, stream>>>(pacc, plsum,
                                                                    out);
}
